// MSDeformAttnPixelDecoder_7773890805840
// MI455X (gfx1250) — matrix operand order repaired and checked by reading, not compiled, not run on hardware
//
#include <hip/hip_runtime.h>
#include <hip/hip_bf16.h>
#include <math.h>

// ---------------------------------------------------------------------------
// Deformable-DETR encoder for MI455X (gfx1250, wave32).
//  * GEMMs: v_wmma_f32_16x16x32_bf16; bf16 A staged via global_load_async_
//    to_lds_b128; bf16 weights pre-transposed [N,K] -> B frag = one 32B load;
//    32-row M tile so each B fragment feeds two WMMAs.
//  * FFN fused 256->1024->256; 16x1024 hidden tile lives in LDS.
//  * sampling: one wave per (b,s,h), lane = head-dim channel.
// ---------------------------------------------------------------------------

#define B_  2
#define S_  20197
#define C_  256
#define NH_ 8
#define HD_ 32
#define NL_ 4
#define NP_ 4
#define DFF_ 1024
#define NLAYER_ 6
#define BS_ (B_ * S_)   // 40394

#define APAD_   8                 // LDS row padding (bf16 elems) -> no bank conflicts
#define ASTRIDE (C_ + APAD_)      // 264
#define HSTRIDE (DFF_ + APAD_)    // 1032

typedef __attribute__((ext_vector_type(16))) __bf16          v16bf;
typedef __attribute__((ext_vector_type(16))) unsigned short  u16x16;
typedef __attribute__((ext_vector_type(8)))  unsigned short  u16x8;
typedef __attribute__((ext_vector_type(8)))  float           v8f;
typedef int i32x4 __attribute__((vector_size(16)));          // matches builtin proto

#if __has_builtin(__builtin_amdgcn_global_load_async_to_lds_b128) && \
    __has_builtin(__builtin_amdgcn_s_wait_asynccnt)
#define HAVE_ASYNC_LDS 1
#else
#define HAVE_ASYNC_LDS 0
#endif

// ---- f32 -> bf16 (RNE) -----------------------------------------------------
__device__ __forceinline__ unsigned short f2bf(float f) {
    unsigned u = __builtin_bit_cast(unsigned, f);
    u = (u + 0x7FFFu + ((u >> 16) & 1u)) >> 16;
    return (unsigned short)u;
}

// ---- 16B global -> LDS copy (async path when available) --------------------
__device__ __forceinline__ void g2l_b128(const unsigned short* g, unsigned short* l) {
#if HAVE_ASYNC_LDS
    __builtin_amdgcn_global_load_async_to_lds_b128(
        (__attribute__((address_space(1))) i32x4*)g,
        (__attribute__((address_space(3))) i32x4*)l, 0, 0);
#else
    *(u16x8*)l = *(const u16x8*)g;
#endif
}

__device__ __forceinline__ void stage_fence() {
#if HAVE_ASYNC_LDS
    __builtin_amdgcn_s_wait_asynccnt(0);
#endif
    __syncthreads();
}

// ---- A fragment 16x32 bf16 from LDS ----------------------------------------
__device__ __forceinline__ v16bf load_a_frag(const unsigned short* p) {
    u16x8 lo = *(const u16x8*)p;
    u16x8 hi = *(const u16x8*)(p + 8);
    u16x16 t = __builtin_shufflevector(lo, hi, 0, 1, 2, 3, 4, 5, 6, 7,
                                               8, 9, 10, 11, 12, 13, 14, 15);
    return __builtin_bit_cast(v16bf, t);
}

// ---- B fragment 32x16 bf16 from global WT[N,K] -----------------------------
__device__ __forceinline__ v16bf load_b_frag(const unsigned short* p) {
    u16x16 t = *(const u16x16*)p;
    return __builtin_bit_cast(v16bf, t);
}

// ===========================================================================
// Weight pack: W[K,N] f32 -> WT[N,K] bf16 (once per launch).
// ===========================================================================
__global__ void pack_wt_kernel(const float* __restrict__ W,
                               unsigned short* __restrict__ WT, int K, int N) {
    int i = blockIdx.x * blockDim.x + threadIdx.x;
    if (i < K * N) {
        int k = i / N, n = i - k * N;
        WT[(size_t)n * K + k] = f2bf(W[i]);
    }
}

// ===========================================================================
// q = x + pos -> bf16; also emit bf16 copy of x.
// ===========================================================================
__global__ void add_pos_kernel(const float* __restrict__ x,
                               const float* __restrict__ pos,
                               unsigned short* __restrict__ qbf,
                               unsigned short* __restrict__ xbf, int n) {
    int i = blockIdx.x * blockDim.x + threadIdx.x;
    if (i < n) {
        float xv = x[i];
        qbf[i] = f2bf(xv + pos[i]);
        xbf[i] = f2bf(xv);
    }
}

// ===========================================================================
// GEMM: out[M,N] = A_bf16[M,256] @ WT_bf16[N,256]^T + bias.
// Block = 128 threads (4 waves); block tile 32(M) x 64(N); each wave: one
// 16-wide N tile x two 16-row M subtiles -> every B fragment feeds 2 WMMAs.
// N is compile-time -> store addressing is shifts, not v_mul_u64.
// ===========================================================================
template <int N>
__global__ void gemm_kernel(const unsigned short* __restrict__ A,
                            const unsigned short* __restrict__ WT,
                            const float* __restrict__ bias,
                            float* __restrict__ out, int M) {
    __shared__ unsigned short sA[32 * ASTRIDE];
    const int m0   = blockIdx.x * 32;
    const int tid  = threadIdx.x;
    const int lane = tid & 31;
    const int wave = tid >> 5;

    // Stage 32x256 bf16 (16KB): 1024 x 16B chunks, 8 per thread, async-to-LDS.
#pragma unroll
    for (int it = 0; it < 8; ++it) {
        int c    = tid + it * 128;
        int row  = c >> 5;            // 32 chunks per 512B row
        int cole = (c & 31) * 8;      // bf16 element offset in row
        int rowc = m0 + row; if (rowc >= M) rowc = M - 1;   // clamp (junk rows unstored)
        g2l_b128(A + (size_t)rowc * C_ + cole, sA + row * ASTRIDE + cole);
    }
    stage_fence();

    const int n0    = blockIdx.y * 64 + wave * 16;
    const int nlane = lane & 15;
    const int khi   = lane >> 4;
    const unsigned short* ab0 = sA + nlane * ASTRIDE + khi * 16;         // rows 0-15
    const unsigned short* ab1 = ab0 + 16 * ASTRIDE;                      // rows 16-31
    const unsigned short* bb  = WT + (size_t)(n0 + nlane) * C_ + khi * 16;

    v8f acc0 = {0.f, 0.f, 0.f, 0.f, 0.f, 0.f, 0.f, 0.f};
    v8f acc1 = {0.f, 0.f, 0.f, 0.f, 0.f, 0.f, 0.f, 0.f};
#pragma unroll
    for (int k0 = 0; k0 < C_; k0 += 32) {
        v16bf b  = load_b_frag(bb + k0);
        v16bf a0 = load_a_frag(ab0 + k0);
        acc0 = __builtin_amdgcn_wmma_f32_16x16x32_bf16(
            false, a0, false, b, (short)0, acc0, false, false);
        v16bf a1 = load_a_frag(ab1 + k0);
        acc1 = __builtin_amdgcn_wmma_f32_16x16x32_bf16(
            false, a1, false, b, (short)0, acc1, false, false);
    }

    const int   n  = n0 + nlane;
    const float bs = bias[n];
#pragma unroll
    for (int half = 0; half < 2; ++half) {
        const v8f& acc   = half ? acc1 : acc0;
        const int  mbase = m0 + half * 16 + khi * 8;
        if (m0 + half * 16 + 16 <= M) {           // fast path
            float* o = out + (size_t)mbase * N + n;
#pragma unroll
            for (int v = 0; v < 8; ++v) o[(size_t)v * N] = acc[v] + bs;
        } else {
#pragma unroll
            for (int v = 0; v < 8; ++v)
                if (mbase + v < M) out[(size_t)(mbase + v) * N + n] = acc[v] + bs;
        }
    }
}

// ===========================================================================
// Fused FFN: out = relu(x @ W1 + b1) @ W2 + b2; hidden 16x1024 bf16 in LDS.
// Block = 256 threads (8 waves), 16-row M tile.
// ===========================================================================
__global__ void ffn_kernel(const unsigned short* __restrict__ xbf,
                           const unsigned short* __restrict__ W1T,  // [DFF, C]
                           const float* __restrict__ b1,
                           const unsigned short* __restrict__ W2T,  // [C, DFF]
                           const float* __restrict__ b2,
                           float* __restrict__ out, int M) {
    __shared__ unsigned short sA[16 * ASTRIDE];
    __shared__ unsigned short sH[16 * HSTRIDE];
    const int m0   = blockIdx.x * 16;
    const int tid  = threadIdx.x;
    const int lane = tid & 31;
    const int wave = tid >> 5;

#pragma unroll
    for (int it = 0; it < 2; ++it) {
        int c    = tid + it * 256;
        int row  = c >> 5;
        int cole = (c & 31) * 8;
        int rowc = m0 + row; if (rowc >= M) rowc = M - 1;
        g2l_b128(xbf + (size_t)rowc * C_ + cole, sA + row * ASTRIDE + cole);
    }
    stage_fence();

    const int nlane = lane & 15;
    const int khi   = lane >> 4;
    const int mloc  = khi * 8;
    const unsigned short* ab = sA + nlane * ASTRIDE + khi * 16;

    // Phase 1: hidden = relu(x @ W1 + b1) -> LDS.  64 n-tiles / 8 waves.
#pragma unroll 1
    for (int t = 0; t < 8; ++t) {
        const int n0 = (t * 8 + wave) * 16;
        const unsigned short* bb = W1T + (size_t)(n0 + nlane) * C_ + khi * 16;
        v8f acc = {0.f, 0.f, 0.f, 0.f, 0.f, 0.f, 0.f, 0.f};
#pragma unroll
        for (int k0 = 0; k0 < C_; k0 += 32) {
            v16bf a = load_a_frag(ab + k0);
            v16bf b = load_b_frag(bb + k0);
            acc = __builtin_amdgcn_wmma_f32_16x16x32_bf16(
                false, a, false, b, (short)0, acc, false, false);
        }
        const float bs = b1[n0 + nlane];
#pragma unroll
        for (int v = 0; v < 8; ++v) {
            float h = acc[v] + bs;
            h = h > 0.f ? h : 0.f;
            sH[(mloc + v) * HSTRIDE + n0 + nlane] = f2bf(h);
        }
    }
    __syncthreads();

    // Phase 2: out = hidden @ W2 + b2.  16 n-tiles / 8 waves = 2 each.
    const unsigned short* hb = sH + nlane * HSTRIDE + khi * 16;
#pragma unroll 1
    for (int t = 0; t < 2; ++t) {
        const int n0 = (wave * 2 + t) * 16;
        const unsigned short* bb = W2T + (size_t)(n0 + nlane) * DFF_ + khi * 16;
        v8f acc = {0.f, 0.f, 0.f, 0.f, 0.f, 0.f, 0.f, 0.f};
#pragma unroll 4
        for (int k0 = 0; k0 < DFF_; k0 += 32) {
            v16bf a = load_a_frag(hb + k0);
            v16bf b = load_b_frag(bb + k0);
            acc = __builtin_amdgcn_wmma_f32_16x16x32_bf16(
                false, a, false, b, (short)0, acc, false, false);
        }
        const int   n     = n0 + nlane;
        const int   mbase = m0 + mloc;
        const float bs    = b2[n];
        if (m0 + 16 <= M) {
            float* o = out + (size_t)mbase * C_ + n;
#pragma unroll
            for (int v = 0; v < 8; ++v) o[(size_t)v * C_] = acc[v] + bs;
        } else {
#pragma unroll
            for (int v = 0; v < 8; ++v)
                if (mbase + v < M) out[(size_t)(mbase + v) * C_ + n] = acc[v] + bs;
        }
    }
}

// ===========================================================================
// Multi-scale deformable sampling + softmax -> bf16 output (GEMM input).
// ===========================================================================
__global__ void msda_sample_kernel(const float* __restrict__ val,
                                   const float* __restrict__ off,
                                   const float* __restrict__ attlog,
                                   unsigned short* __restrict__ out) {
    const int LVLH[NL_]   = {100, 50, 25, 13};
    const int LVLW[NL_]   = {152, 76, 38, 19};
    const int LSTART[NL_] = {0, 15200, 19000, 19950};

    const int wid  = blockIdx.x * (blockDim.x >> 5) + (threadIdx.x >> 5);
    const int lane = threadIdx.x & 31;
    if (wid >= BS_ * NH_) return;

    const int b   = wid / (S_ * NH_);
    const int rem = wid - b * (S_ * NH_);
    const int s   = rem / NH_;
    const int h   = rem - s * NH_;

    int lvl = 3;
#pragma unroll
    for (int l = NL_ - 1; l >= 0; --l)
        if (s >= LSTART[l]) { lvl = l; break; }
    const int j  = s - LSTART[lvl];
    const int iy = j / LVLW[lvl];
    const int ix = j - iy * LVLW[lvl];
    const float refx = (ix + 0.5f) / (float)LVLW[lvl];
    const float refy = (iy + 0.5f) / (float)LVLH[lvl];

    const float* al = attlog + (size_t)(b * S_ + s) * (NH_ * NL_ * NP_) + h * 16;
    float a[16], mx = -1e30f;
#pragma unroll
    for (int i = 0; i < 16; ++i) { a[i] = al[i]; mx = fmaxf(mx, a[i]); }
    float sum = 0.f;
#pragma unroll
    for (int i = 0; i < 16; ++i) { a[i] = __expf(a[i] - mx); sum += a[i]; }
    const float inv = 1.f / sum;

    const float* ob = off + (size_t)(b * S_ + s) * (NH_ * NL_ * NP_ * 2) + h * 32;
    float acc = 0.f;
#pragma unroll 1
    for (int l = 0; l < NL_; ++l) {
        const int Hl = LVLH[l], Wl = LVLW[l], st = LSTART[l];
        const float fW = (float)Wl, fH = (float)Hl;
#pragma unroll 1
        for (int p = 0; p < NP_; ++p) {
            const float ox = ob[(l * NP_ + p) * 2 + 0];
            const float oy = ob[(l * NP_ + p) * 2 + 1];
            const float lx = (refx + ox / fW) * fW - 0.5f;
            const float ly = (refy + oy / fH) * fH - 0.5f;
            const float x0 = floorf(lx), y0 = floorf(ly);
            const float fx = lx - x0, fy = ly - y0;
            const int x0i = (int)x0, y0i = (int)y0;
            const float aw = a[l * NP_ + p] * inv;
            float samp = 0.f;
#pragma unroll
            for (int dy = 0; dy < 2; ++dy) {
#pragma unroll
                for (int dx = 0; dx < 2; ++dx) {
                    const int xi = x0i + dx, yi = y0i + dy;
                    const float wx = dx ? fx : 1.f - fx;
                    const float wy = dy ? fy : 1.f - fy;
                    if (xi >= 0 && xi < Wl && yi >= 0 && yi < Hl) {
                        const int idx = st + yi * Wl + xi;
                        samp += wx * wy *
                            val[((size_t)(b * S_ + idx) * NH_ + h) * HD_ + lane];
                    }
                }
            }
            acc += aw * samp;
        }
    }
    out[((size_t)(b * S_ + s) * NH_ + h) * HD_ + lane] = f2bf(acc);
}

// ===========================================================================
// out = LayerNorm(x + y) * g + b (f32), optional bf16 mirror.
// ===========================================================================
__global__ void ln_residual_kernel(const float* __restrict__ x,
                                   const float* __restrict__ y,
                                   const float* __restrict__ g,
                                   const float* __restrict__ b,
                                   float* __restrict__ out,
                                   unsigned short* __restrict__ obf, int rows) {
    const int wid  = blockIdx.x * (blockDim.x >> 5) + (threadIdx.x >> 5);
    const int lane = threadIdx.x & 31;
    if (wid >= rows) return;
    const float* xr = x + (size_t)wid * C_;
    const float* yr = y + (size_t)wid * C_;
    float v[8], s = 0.f;
#pragma unroll
    for (int i = 0; i < 8; ++i) {
        int c = lane + 32 * i;
        v[i] = xr[c] + yr[c];
        s += v[i];
    }
#pragma unroll
    for (int o = 16; o; o >>= 1) s += __shfl_xor(s, o, 32);
    const float mean = s * (1.f / C_);
    float var = 0.f;
#pragma unroll
    for (int i = 0; i < 8; ++i) { float d = v[i] - mean; var += d * d; }
#pragma unroll
    for (int o = 16; o; o >>= 1) var += __shfl_xor(var, o, 32);
    const float inv = rsqrtf(var * (1.f / C_) + 1e-5f);
    float* orow = out + (size_t)wid * C_;
#pragma unroll
    for (int i = 0; i < 8; ++i) {
        int c = lane + 32 * i;
        float r = (v[i] - mean) * inv * g[c] + b[c];
        orow[c] = r;
        if (obf) obf[(size_t)wid * C_ + c] = f2bf(r);
    }
}

// ===========================================================================
// Host driver
// ===========================================================================
extern "C" void kernel_launch(void* const* d_in, const int* in_sizes, int n_in,
                              void* d_out, int out_size, void* d_ws, size_t ws_size,
                              hipStream_t stream) {
    const float* src  = (const float*)d_in[0];
    const float* pos  = (const float*)d_in[1];
    const float* Woff = (const float*)d_in[2];
    const float* boff = (const float*)d_in[3];
    const float* Watt = (const float*)d_in[4];
    const float* batt = (const float*)d_in[5];
    const float* Wval = (const float*)d_in[6];
    const float* bval = (const float*)d_in[7];
    const float* Wout = (const float*)d_in[8];
    const float* bout = (const float*)d_in[9];
    const float* W1   = (const float*)d_in[10];
    const float* b1   = (const float*)d_in[11];
    const float* W2   = (const float*)d_in[12];
    const float* b2   = (const float*)d_in[13];
    const float* g1   = (const float*)d_in[14];
    const float* be1  = (const float*)d_in[15];
    const float* g2   = (const float*)d_in[16];
    const float* be2  = (const float*)d_in[17];

    const int M = BS_;
    // ---- workspace carve-up ----
    float* ws   = (float*)d_ws;
    float* val  = ws;
    float* off  = val + (size_t)M * C_;
    float* attl = off + (size_t)M * 256;
    float* tmp  = attl + (size_t)M * 128;
    float* xbuf = tmp + (size_t)M * C_;
    unsigned short* qbf    = (unsigned short*)(xbuf + (size_t)M * C_);
    unsigned short* xbf    = qbf + (size_t)M * C_;
    unsigned short* msdabf = xbf + (size_t)M * C_;
    unsigned short* wt     = msdabf + (size_t)M * C_;

    // per-layer packed-weight offsets (bf16 elems)
    const size_t oVal = 0;
    const size_t oOff = oVal + 256 * 256;
    const size_t oAtt = oOff + 256 * 256;
    const size_t oOut = oAtt + 128 * 256;
    const size_t oW1  = oOut + 256 * 256;
    const size_t oW2  = oW1 + 1024 * 256;
    const size_t wtLayer = oW2 + 1024 * 256;   // 753664

    // ---- pack all weights once: f32 [K,N] -> bf16 [N,K] ----
    for (int l = 0; l < NLAYER_; ++l) {
        unsigned short* wl = wt + (size_t)l * wtLayer;
        pack_wt_kernel<<<(256 * 256 + 255) / 256, 256, 0, stream>>>(
            Wval + (size_t)l * 256 * 256, wl + oVal, 256, 256);
        pack_wt_kernel<<<(256 * 256 + 255) / 256, 256, 0, stream>>>(
            Woff + (size_t)l * 256 * 256, wl + oOff, 256, 256);
        pack_wt_kernel<<<(256 * 128 + 255) / 256, 256, 0, stream>>>(
            Watt + (size_t)l * 256 * 128, wl + oAtt, 256, 128);
        pack_wt_kernel<<<(256 * 256 + 255) / 256, 256, 0, stream>>>(
            Wout + (size_t)l * 256 * 256, wl + oOut, 256, 256);
        pack_wt_kernel<<<(256 * 1024 + 255) / 256, 256, 0, stream>>>(
            W1 + (size_t)l * 256 * 1024, wl + oW1, 256, 1024);
        pack_wt_kernel<<<(1024 * 256 + 255) / 256, 256, 0, stream>>>(
            W2 + (size_t)l * 1024 * 256, wl + oW2, 1024, 256);
    }

    const int m32Blocks = (M + 31) / 32;         // 1263
    const int m16Blocks = (M + 15) / 16;         // 2525
    const dim3 g256(m32Blocks, 4);               // N = 256
    const dim3 g128(m32Blocks, 2);               // N = 128
    const int lnBlocks  = (M + 7) / 8;
    const int smpBlocks = (M * NH_ + 7) / 8;
    const int ewBlocks  = (M * C_ + 255) / 256;

    const float* xin = src;
    for (int l = 0; l < NLAYER_; ++l) {
        unsigned short* wl = wt + (size_t)l * wtLayer;

        add_pos_kernel<<<ewBlocks, 256, 0, stream>>>(xin, pos, qbf, xbf, M * C_);

        gemm_kernel<256><<<g256, 128, 0, stream>>>(xbf, wl + oVal, bval + l * C_,
                                                   val, M);
        gemm_kernel<256><<<g256, 128, 0, stream>>>(qbf, wl + oOff, boff + l * 256,
                                                   off, M);
        gemm_kernel<128><<<g128, 128, 0, stream>>>(qbf, wl + oAtt, batt + l * 128,
                                                   attl, M);

        msda_sample_kernel<<<smpBlocks, 256, 0, stream>>>(val, off, attl, msdabf);

        gemm_kernel<256><<<g256, 128, 0, stream>>>(msdabf, wl + oOut, bout + l * C_,
                                                   tmp, M);

        // LN1 also emits bf16 for the FFN (reuses qbf — dead after Woff/Watt)
        ln_residual_kernel<<<lnBlocks, 256, 0, stream>>>(
            xin, tmp, g1 + l * C_, be1 + l * C_, xbuf, qbf, M);

        ffn_kernel<<<m16Blocks, 256, 0, stream>>>(
            qbf, wl + oW1, b1 + l * DFF_, wl + oW2, b2 + l * C_, tmp, M);

        float* xdst = (l == NLAYER_ - 1) ? (float*)d_out : xbuf;
        ln_residual_kernel<<<lnBlocks, 256, 0, stream>>>(
            xbuf, tmp, g2 + l * C_, be2 + l * C_, xdst, nullptr, M);

        xin = xbuf;
    }
}
